// CTCModel_80324478369871
// MI455X (gfx1250) — compile-verified, matrix operand order checked by reading, not run on hardware
//
#include <hip/hip_runtime.h>
#include <math.h>

#define Bn 32
#define Tn 512
#define Fn 120
#define Hn 512
#define Cn 62
#define Ln 80
#define Sn (2 * Ln + 1) /* 161 */
#define BLANKC (Cn - 1)
#define NEGV (-1e30f)
#define Kpad1 128 /* F=120 padded */
#define Cpad 64   /* C=62 padded  */

typedef __attribute__((ext_vector_type(16))) _Float16 v16h;
typedef __attribute__((ext_vector_type(8)))  _Float16 v8h;
typedef __attribute__((ext_vector_type(8)))  float    v8f;

#if defined(__has_builtin)
#if __has_builtin(__builtin_amdgcn_tensor_load_to_lds) && \
    __has_builtin(__builtin_amdgcn_s_wait_tensorcnt)
#define HAVE_TDM 1
#endif
#endif
#ifndef HAVE_TDM
#define HAVE_TDM 0
#endif

// K-offset pattern for 16-bit A/B fragments (16x32 / 32x16):
// lanes 0-15 hold K {0..7, 16..23}; lanes 16-31 hold K {8..15, 24..31}.
__device__ __forceinline__ int kofs(int lane, int j) {
    int lo = (lane & 16) ? 8 : 0;
    return (j < 8) ? (lo + j) : (16 + lo + (j - 8));
}

__device__ __forceinline__ v8f wmma32(v16h a, v16h b, v8f c) {
    return __builtin_amdgcn_wmma_f32_16x16x32_f16(false, a, false, b, (short)0, c,
                                                  false, false);
}

// A fragment from f32 row (global or LDS): two contiguous 8-float runs.
__device__ __forceinline__ v16h load_a_f32(const float* xr, int k0, int lo) {
    const float4* p0 = (const float4*)(xr + k0 + lo);
    const float4* p1 = (const float4*)(xr + k0 + 16 + lo);
    float4 x0 = p0[0], x1 = p0[1], x2 = p1[0], x3 = p1[1];
    v16h a;
    auto c0 = __builtin_amdgcn_cvt_pkrtz(x0.x, x0.y);
    auto c1 = __builtin_amdgcn_cvt_pkrtz(x0.z, x0.w);
    auto c2 = __builtin_amdgcn_cvt_pkrtz(x1.x, x1.y);
    auto c3 = __builtin_amdgcn_cvt_pkrtz(x1.z, x1.w);
    auto c4 = __builtin_amdgcn_cvt_pkrtz(x2.x, x2.y);
    auto c5 = __builtin_amdgcn_cvt_pkrtz(x2.z, x2.w);
    auto c6 = __builtin_amdgcn_cvt_pkrtz(x3.x, x3.y);
    auto c7 = __builtin_amdgcn_cvt_pkrtz(x3.z, x3.w);
    a[0] = c0[0]; a[1] = c0[1]; a[2] = c1[0]; a[3] = c1[1];
    a[4] = c2[0]; a[5] = c2[1]; a[6] = c3[0]; a[7] = c3[1];
    a[8] = c4[0]; a[9] = c4[1]; a[10] = c5[0]; a[11] = c5[1];
    a[12] = c6[0]; a[13] = c6[1]; a[14] = c7[0]; a[15] = c7[1];
    return a;
}

// A fragment = sum of two f32 rows (fwd + bwd GRU outputs).
__device__ __forceinline__ v16h load_a_sum(const float* __restrict__ fr,
                                           const float* __restrict__ gr, int k0, int lo) {
    const float4* f0 = (const float4*)(fr + k0 + lo);
    const float4* f1 = (const float4*)(fr + k0 + 16 + lo);
    const float4* g0 = (const float4*)(gr + k0 + lo);
    const float4* g1 = (const float4*)(gr + k0 + 16 + lo);
    float4 x0 = f0[0], x1 = f0[1], x2 = f1[0], x3 = f1[1];
    float4 y0 = g0[0], y1 = g0[1], y2 = g1[0], y3 = g1[1];
    v16h a;
    auto c0 = __builtin_amdgcn_cvt_pkrtz(x0.x + y0.x, x0.y + y0.y);
    auto c1 = __builtin_amdgcn_cvt_pkrtz(x0.z + y0.z, x0.w + y0.w);
    auto c2 = __builtin_amdgcn_cvt_pkrtz(x1.x + y1.x, x1.y + y1.y);
    auto c3 = __builtin_amdgcn_cvt_pkrtz(x1.z + y1.z, x1.w + y1.w);
    auto c4 = __builtin_amdgcn_cvt_pkrtz(x2.x + y2.x, x2.y + y2.y);
    auto c5 = __builtin_amdgcn_cvt_pkrtz(x2.z + y2.z, x2.w + y2.w);
    auto c6 = __builtin_amdgcn_cvt_pkrtz(x3.x + y3.x, x3.y + y3.y);
    auto c7 = __builtin_amdgcn_cvt_pkrtz(x3.z + y3.z, x3.w + y3.w);
    a[0] = c0[0]; a[1] = c0[1]; a[2] = c1[0]; a[3] = c1[1];
    a[4] = c2[0]; a[5] = c2[1]; a[6] = c3[0]; a[7] = c3[1];
    a[8] = c4[0]; a[9] = c4[1]; a[10] = c5[0]; a[11] = c5[1];
    a[12] = c6[0]; a[13] = c6[1]; a[14] = c7[0]; a[15] = c7[1];
    return a;
}

// A fragment from f16 LDS row: two 16-byte ds_load_b128.
__device__ __forceinline__ v16h load_a_h16(const _Float16* hr, int k0, int lo) {
    v8h u0 = *(const v8h*)(hr + k0 + lo);
    v8h u1 = *(const v8h*)(hr + k0 + 16 + lo);
    v16h a;
#pragma unroll
    for (int j = 0; j < 8; ++j) { a[j] = u0[j]; a[8 + j] = u1[j]; }
    return a;
}

#if HAVE_TDM
// Issue a TDM 2D tile load: [Bn rows x Hn f32], row stride Tn*Hn elems,
// from global gsrc into LDS byte address lds_byte_addr. (D# per ISA ch.8)
// This toolchain's builtin arity is 6: (g0, g1, g2, g3, g4, cpol).
__device__ __forceinline__ void tdm_load_x_tile(const float* gsrc, unsigned lds_byte_addr) {
    typedef unsigned int u32x4 __attribute__((ext_vector_type(4)));
    typedef int i32x4v __attribute__((ext_vector_type(4)));
    typedef int i32x8v __attribute__((ext_vector_type(8)));
    unsigned long long ga = (unsigned long long)(uintptr_t)gsrc;
    u32x4 g0;
    g0[0] = 1u;                                               // count=1 (valid)
    g0[1] = lds_byte_addr;                                    // lds_addr
    g0[2] = (unsigned)(ga & 0xffffffffu);                     // global_addr[31:0]
    g0[3] = (unsigned)((ga >> 32) & 0x1ffffffull) | (2u << 30); // [56:32] | type=2
    i32x8v g1;
    g1[0] = (2 << 16);       // workgroup_mask=0, data_size=4B
    g1[1] = (int)(Hn << 16); // tensor_dim0[15:0]
    g1[2] = (int)(Bn << 16); // tensor_dim0[31:16]=0, tensor_dim1[15:0]=Bn
    g1[3] = (int)(Hn << 16); // tensor_dim1[31:16]=0, tile_dim0=Hn
    g1[4] = Bn;              // tile_dim1=Bn, tile_dim2=0
    g1[5] = Tn * Hn;         // tensor_dim0_stride[31:0]
    g1[6] = 0;               // stride hi / dim1_stride lo
    g1[7] = 0;
    i32x4v z4 = {0, 0, 0, 0};
    i32x8v z8 = {0, 0, 0, 0, 0, 0, 0, 0};
    __builtin_amdgcn_tensor_load_to_lds(g0, g1, z4, z4, z8, 0);
}
#endif

// ---------------------------------------------------------------------------
// One-time packers (zero-padded, fragment-native f16 weight layout
// [ntile][kstep][lane][16] so B operand = one contiguous 32B load per lane).
// ---------------------------------------------------------------------------
__global__ void pad_inputs_kernel(const float* __restrict__ X, float* __restrict__ Xp) {
    int idx = blockIdx.x * blockDim.x + threadIdx.x;
    if (idx >= Bn * Tn * Kpad1) return;
    int k = idx & (Kpad1 - 1);
    int m = idx >> 7;
    Xp[idx] = (k < Fn) ? X[(size_t)m * Fn + k] : 0.f;
}

__global__ void pack_weight_f16(const float* __restrict__ W, _Float16* __restrict__ Wp,
                                int Ksrc, int Nsrc, int Kp, int Np) {
    size_t idx = (size_t)blockIdx.x * blockDim.x + threadIdx.x;
    int ksteps = Kp >> 5;
    size_t total = (size_t)(Np >> 4) * ksteps * 32 * 16;
    if (idx >= total) return;
    int j = idx & 15;
    int lane = (idx >> 4) & 31;
    size_t rest = idx >> 9;
    int ks = (int)(rest % ksteps);
    int nt = (int)(rest / ksteps);
    int k = (ks << 5) + kofs(lane, j);
    int n = (nt << 4) + (lane & 15);
    float v = (k < Ksrc && n < Nsrc) ? W[(size_t)k * Nsrc + n] : 0.f;
    Wp[idx] = (_Float16)v;
}

// ---------------------------------------------------------------------------
// Y = act(X[MxK] @ Wp + bias). K mult of 32, N mult of 16. No inner branches.
// ---------------------------------------------------------------------------
__global__ void gemm_f16packed(const float* __restrict__ X, const _Float16* __restrict__ Wp,
                               const float* __restrict__ bias, float* __restrict__ Y,
                               int M, int N, int K, int relu) {
    int lane = threadIdx.x & 31;
    int wg   = blockIdx.x * (blockDim.x >> 5) + (threadIdx.x >> 5);
    int mtiles = M >> 4, ntiles = N >> 4;
    if (wg >= mtiles * ntiles) return;
    int mbase = (wg % mtiles) << 4;
    int nbase = (wg / mtiles) << 4;
    int row = mbase + (lane & 15);
    int col = nbase + (lane & 15);
    int lo = (lane & 16) ? 8 : 0;
    int ksteps = K >> 5;
    const float* xr = X + (size_t)row * K;
    const _Float16* wp = Wp + ((size_t)(nbase >> 4) * ksteps * 32 + lane) * 16;
    v8f acc = {};
#pragma unroll 2
    for (int ks = 0; ks < ksteps; ++ks) {
        __builtin_prefetch((const void*)(wp + (size_t)(ks + 1) * 512), 0, 1);
        v16h a = load_a_f32(xr, ks << 5, lo);
        v16h b = *(const v16h*)(wp + (size_t)ks * 512);
        acc = wmma32(a, b, acc);
    }
    int mo = (lane & 16) ? 8 : 0;
    float bv = bias[col];
#pragma unroll
    for (int i = 0; i < 8; ++i) {
        float v = acc[i] + bv;
        if (relu) v = fmaxf(v, 0.f);
        Y[(size_t)(mbase + mo + i) * N + col] = v;
    }
}

// ---------------------------------------------------------------------------
// Persistent BiGRU: block 0 = fwd, block 1 = bwd (reversal by indexing).
// Per step, x_t is staged into LDS: fwd via Tensor Data Mover (2D tile DMA),
// bwd via vectorized cooperative copy (per-row reversed gather). h kept in
// f32 LDS (state) + f16 LDS mirror (A operand); r*h f16; u f32.
// ---------------------------------------------------------------------------
__global__ void gru_kernel(const float* __restrict__ x,
                           const _Float16* fWgp, const float* fbg,
                           const _Float16* fWcp, const float* fbc,
                           const _Float16* bWgp, const float* bbg,
                           const _Float16* bWcp, const float* bbc,
                           const int* __restrict__ lens,
                           float* __restrict__ out_f, float* __restrict__ out_b) {
    extern __shared__ char smem[];
    float*    h32 = (float*)smem;                                 // 64KB
    _Float16* h16 = (_Float16*)(smem + Bn * Hn * 4);              // 32KB
    _Float16* rh  = (_Float16*)(smem + Bn * Hn * 6);              // 32KB
    float*    uld = (float*)(smem + Bn * Hn * 8);                 // 64KB
    float*    xst = (float*)(smem + Bn * Hn * 12);                // 64KB x_t stage
    int*      lld = (int*)(smem + Bn * Hn * 16);                  // 128B

    int dir = blockIdx.x;
    const _Float16* Wgp = dir ? bWgp : fWgp;
    const float*    bg  = dir ? bbg : fbg;
    const _Float16* Wcp = dir ? bWcp : fWcp;
    const float*    bc  = dir ? bbc : fbc;
    float* out = dir ? out_b : out_f;
    int rev = dir;

    int tid  = threadIdx.x;
    int lane = tid & 31;
    int wave = tid >> 5; // 0..15
    int lo   = (lane & 16) ? 8 : 0;
    int mo   = (lane & 16) ? 8 : 0;

    for (int i = tid; i < Bn * Hn; i += blockDim.x) { h32[i] = 0.f; h16[i] = (_Float16)0.f; }
    if (tid < Bn) lld[tid] = lens[tid];
    __syncthreads();

#pragma unroll 1
    for (int t = 0; t < Tn; ++t) {
        // ---- stage x_t[32x512] into LDS ----
        if (rev == 0) {
#if HAVE_TDM
            if (wave == 0) {
                tdm_load_x_tile(x + (size_t)t * Hn, (unsigned)(uintptr_t)xst);
                __builtin_amdgcn_s_wait_tensorcnt(0);
            }
#else
#pragma unroll 1
            for (int q = 0; q < 8; ++q) {
                int fidx = tid + q * 512;
                int row = fidx >> 7, c4 = fidx & 127;
                ((float4*)(xst + row * Hn))[c4] =
                    ((const float4*)(x + ((size_t)row * Tn + t) * Hn))[c4];
            }
#endif
        } else {
#pragma unroll 1
            for (int q = 0; q < 8; ++q) {
                int fidx = tid + q * 512;
                int row = fidx >> 7, c4 = fidx & 127;
                int lm = lld[row];
                int tin = (t < lm) ? (lm - 1 - t) : t;
                ((float4*)(xst + row * Hn))[c4] =
                    ((const float4*)(x + ((size_t)row * Tn + tin) * Hn))[c4];
            }
        }
        __syncthreads();

        // ---- gates: sigmoid([x_t ; h] @ Wg + bg) -> r*h (f16), u (f32) ----
#pragma unroll 1
        for (int ti = 0; ti < 8; ++ti) {
            int tg = wave * 8 + ti;      // 0..127
            int mbase = (tg & 1) << 4;
            int ntile = tg >> 1;         // 0..63
            int nbase = ntile << 4;
            int arow = mbase + (lane & 15);
            const float* xrow = xst + arow * Hn;
            const _Float16* hrow = h16 + arow * Hn;
            const _Float16* wp = Wgp + ((size_t)ntile * 32 * 32 + lane) * 16;
            int bcol = nbase + (lane & 15);
            v8f acc = {};
#pragma unroll 2
            for (int ks = 0; ks < 16; ++ks) { // x part (LDS f32)
                __builtin_prefetch((const void*)(wp + (size_t)(ks + 1) * 512), 0, 1);
                v16h a = load_a_f32(xrow, ks << 5, lo);
                v16h b = *(const v16h*)(wp + (size_t)ks * 512);
                acc = wmma32(a, b, acc);
            }
#pragma unroll 2
            for (int ks = 16; ks < 32; ++ks) { // h part (LDS f16)
                v16h a = load_a_h16(hrow, (ks - 16) << 5, lo);
                v16h b = *(const v16h*)(wp + (size_t)ks * 512);
                acc = wmma32(a, b, acc);
            }
            float bgv = bg[bcol];
#pragma unroll
            for (int i = 0; i < 8; ++i) {
                int m = mbase + mo + i;
                float g = 1.f / (1.f + expf(-(acc[i] + bgv)));
                if (nbase < Hn) {
                    rh[m * Hn + bcol] = (_Float16)(g * h32[m * Hn + bcol]);
                } else {
                    uld[m * Hn + (bcol - Hn)] = g;
                }
            }
        }
        __syncthreads();

        // ---- candidate: tanh([x_t ; r*h] @ Wc + bc) + state update ----
#pragma unroll 1
        for (int ti = 0; ti < 4; ++ti) {
            int tc = wave * 4 + ti;      // 0..63
            int mbase = (tc & 1) << 4;
            int ntile = tc >> 1;         // 0..31
            int nbase = ntile << 4;
            int arow = mbase + (lane & 15);
            const float* xrow = xst + arow * Hn;
            const _Float16* rrow = rh + arow * Hn;
            const _Float16* wp = Wcp + ((size_t)ntile * 32 * 32 + lane) * 16;
            int bcol = nbase + (lane & 15);
            v8f acc = {};
#pragma unroll 2
            for (int ks = 0; ks < 16; ++ks) {
                __builtin_prefetch((const void*)(wp + (size_t)(ks + 1) * 512), 0, 1);
                v16h a = load_a_f32(xrow, ks << 5, lo);
                v16h b = *(const v16h*)(wp + (size_t)ks * 512);
                acc = wmma32(a, b, acc);
            }
#pragma unroll 2
            for (int ks = 16; ks < 32; ++ks) {
                v16h a = load_a_h16(rrow, (ks - 16) << 5, lo);
                v16h b = *(const v16h*)(wp + (size_t)ks * 512);
                acc = wmma32(a, b, acc);
            }
            float bcv = bc[bcol];
#pragma unroll
            for (int i = 0; i < 8; ++i) {
                int m = mbase + mo + i;
                float c = tanhf(acc[i] + bcv);
                float u = uld[m * Hn + bcol];
                float hold = h32[m * Hn + bcol];
                float hn = u * hold + (1.f - u) * c;
                int lmm = lld[m];
                bool valid = t < lmm;
                float newh = valid ? hn : hold;
                h32[m * Hn + bcol] = newh;
                h16[m * Hn + bcol] = (_Float16)newh;
                int tout = rev ? (valid ? (lmm - 1 - t) : t) : t;
                out[((size_t)m * Tn + tout) * Hn + bcol] = valid ? hn : 0.f;
            }
        }
        __syncthreads();
    }
}

// ---------------------------------------------------------------------------
// logits[t,b,c] = (fwd+bwd)[b*T+t,:] @ Wf + bf ; packed Wf padded 62->64.
// ---------------------------------------------------------------------------
__global__ void proj_logits(const float* __restrict__ fwd, const float* __restrict__ bwd,
                            const _Float16* __restrict__ Wfp, const float* __restrict__ bf,
                            float* __restrict__ LP) {
    int lane = threadIdx.x & 31;
    int wg   = blockIdx.x * (blockDim.x >> 5) + (threadIdx.x >> 5);
    const int mtiles = (Bn * Tn) >> 4; // 1024
    if (wg >= mtiles * (Cpad >> 4)) return;
    int mbase = (wg % mtiles) << 4;
    int ntile = wg / mtiles;
    int nbase = ntile << 4;
    int col = nbase + (lane & 15);
    int row = mbase + (lane & 15);
    int lo = (lane & 16) ? 8 : 0;
    const float* fr = fwd + (size_t)row * Hn;
    const float* gr = bwd + (size_t)row * Hn;
    const _Float16* wp = Wfp + ((size_t)ntile * (Hn / 32) * 32 + lane) * 16;
    v8f acc = {};
#pragma unroll 2
    for (int ks = 0; ks < Hn / 32; ++ks) {
        __builtin_prefetch((const void*)(wp + (size_t)(ks + 1) * 512), 0, 1);
        v16h a = load_a_sum(fr, gr, ks << 5, lo);
        v16h b = *(const v16h*)(wp + (size_t)ks * 512);
        acc = wmma32(a, b, acc);
    }
    int mo = (lane & 16) ? 8 : 0;
    if (col < Cn) {
        float bv = bf[col];
#pragma unroll
        for (int i = 0; i < 8; ++i) {
            int m = mbase + mo + i;
            int bidx = m / Tn, tt = m % Tn;
            LP[((size_t)tt * Bn + bidx) * Cn + col] = acc[i] + bv;
        }
    }
}

__global__ void log_softmax_kernel(float* __restrict__ LP) {
    int r = blockIdx.x * blockDim.x + threadIdx.x;
    if (r >= Tn * Bn) return;
    float* p = LP + (size_t)r * Cn;
    float mx = -3.4e38f;
    for (int c = 0; c < Cn; ++c) mx = fmaxf(mx, p[c]);
    float s = 0.f;
    for (int c = 0; c < Cn; ++c) s += expf(p[c] - mx);
    float lse = mx + logf(s);
    for (int c = 0; c < Cn; ++c) p[c] -= lse;
}

__device__ __forceinline__ float lse3(float a, float b, float c) {
    float m = fmaxf(a, fmaxf(b, c));
    if (m <= NEGV * 0.5f) return NEGV;
    return m + logf(expf(a - m) + expf(b - m) + expf(c - m));
}

__global__ void ctc_kernel(const float* __restrict__ LP, const int* __restrict__ labels,
                           const int* __restrict__ seq_lens, const int* __restrict__ lab_lens,
                           float* __restrict__ loss) {
    __shared__ int   ext[Sn];
    __shared__ float alpha[2][Sn];
    int b = blockIdx.x;
    int s = threadIdx.x;
    if (s < Sn) ext[s] = (s & 1) ? labels[b * Ln + (s >> 1)] : BLANKC;
    __syncthreads();
    bool skip = false;
    if (s < Sn && (s & 1)) skip = (ext[s] != ((s >= 2) ? ext[s - 2] : BLANKC));
    int sl = seq_lens[b];
    if (s < Sn)
        alpha[0][s] = (s < 2) ? LP[((size_t)0 * Bn + b) * Cn + ext[s]] : NEGV;
    __syncthreads();
    int cur = 0;
    for (int t = 1; t < Tn; ++t) {
        float nv = NEGV;
        if (s < Sn) {
            float a0 = alpha[cur][s];
            float a1 = (s >= 1) ? alpha[cur][s - 1] : NEGV;
            float a2 = (skip && s >= 2) ? alpha[cur][s - 2] : NEGV;
            float em = LP[((size_t)t * Bn + b) * Cn + ext[s]];
            nv = lse3(a0, a1, a2) + em;
            if (t >= sl) nv = a0;
        }
        __syncthreads();
        if (s < Sn) alpha[cur ^ 1][s] = nv;
        __syncthreads();
        cur ^= 1;
    }
    if (s == 0) {
        int ll = lab_lens[b];
        float e1 = alpha[cur][2 * ll - 1];
        float e2 = alpha[cur][2 * ll];
        float m = fmaxf(e1, e2);
        float l = (m <= NEGV * 0.5f) ? NEGV : m + logf(expf(e1 - m) + expf(e2 - m));
        loss[b] = -l;
    }
}

__global__ void finalize_kernel(const float* W1, const float* b1, const float* W2,
                                const float* b2, const float* fWg, const float* fbg,
                                const float* fWc, const float* fbc, const float* bWg,
                                const float* bbg, const float* bWc, const float* bbc,
                                const float* Wf, const float* bfv,
                                const float* __restrict__ ctc, float* __restrict__ outp) {
    __shared__ float red[256];
    const float* ps[14] = {W1, b1, W2, b2, fWg, fbg, fWc, fbc, bWg, bbg, bWc, bbc, Wf, bfv};
    const int ns[14] = {Fn * Hn, Hn, Hn * Hn, Hn, 4 * Hn * Hn, 2 * Hn, 2 * Hn * Hn, Hn,
                        4 * Hn * Hn, 2 * Hn, 2 * Hn * Hn, Hn, Hn * Cn, Cn};
    float acc = 0.f;
    for (int a = 0; a < 14; ++a) {
        const float* p = ps[a];
        int n = ns[a];
        for (int i = threadIdx.x; i < n; i += blockDim.x) {
            float v = p[i];
            acc += v * v;
        }
    }
    red[threadIdx.x] = acc;
    __syncthreads();
    for (int st = 128; st > 0; st >>= 1) {
        if (threadIdx.x < st) red[threadIdx.x] += red[threadIdx.x + st];
        __syncthreads();
    }
    if (threadIdx.x == 0) {
        float l2 = 0.5f * red[0];
        float sm = 0.f, nv = 0.f;
        for (int bb = 0; bb < Bn; ++bb) {
            float v = ctc[bb];
            if (v < 10000.f) { sm += v; nv += 1.f; }
        }
        outp[0] = 1e-4f * l2 + sm / fmaxf(nv, 1.f);
    }
}

static inline size_t align256(size_t x) { return (x + 255) & ~(size_t)255; }

extern "C" void kernel_launch(void* const* d_in, const int* in_sizes, int n_in,
                              void* d_out, int out_size, void* d_ws, size_t ws_size,
                              hipStream_t stream) {
    (void)in_sizes; (void)n_in; (void)out_size; (void)ws_size;
    const float* inputs = (const float*)d_in[0];
    const float* W1 = (const float*)d_in[1];   const float* b1  = (const float*)d_in[2];
    const float* W2 = (const float*)d_in[3];   const float* b2  = (const float*)d_in[4];
    const float* fWg = (const float*)d_in[5];  const float* fbg = (const float*)d_in[6];
    const float* fWc = (const float*)d_in[7];  const float* fbc = (const float*)d_in[8];
    const float* bWg = (const float*)d_in[9];  const float* bbg = (const float*)d_in[10];
    const float* bWc = (const float*)d_in[11]; const float* bbc = (const float*)d_in[12];
    const float* Wf  = (const float*)d_in[13]; const float* bfv = (const float*)d_in[14];
    const int* labels   = (const int*)d_in[15];
    const int* seq_lens = (const int*)d_in[16];
    const int* lab_lens = (const int*)d_in[17];

    char* ws = (char*)d_ws;
    size_t off = 0;
    float* h1 = (float*)(ws + off); off = align256(off + (size_t)Bn * Tn * Hn * 4);
    float* h2 = (float*)(ws + off); off = align256(off + (size_t)Bn * Tn * Hn * 4);
    float* bwdbuf = (float*)(ws + off); off = align256(off + (size_t)Bn * Tn * Hn * 4);
    float* LP = (float*)(ws + off); off = align256(off + (size_t)Tn * Bn * Cn * 4);
    float* ctcbuf = (float*)(ws + off); off = align256(off + Bn * 4);
    float* Xpad = (float*)(ws + off); off = align256(off + (size_t)Bn * Tn * Kpad1 * 4);
    _Float16* W1p  = (_Float16*)(ws + off); off = align256(off + (size_t)Kpad1 * Hn * 2);
    _Float16* W2p  = (_Float16*)(ws + off); off = align256(off + (size_t)Hn * Hn * 2);
    _Float16* fWgp = (_Float16*)(ws + off); off = align256(off + (size_t)4 * Hn * Hn * 2);
    _Float16* fWcp = (_Float16*)(ws + off); off = align256(off + (size_t)2 * Hn * Hn * 2);
    _Float16* bWgp = (_Float16*)(ws + off); off = align256(off + (size_t)4 * Hn * Hn * 2);
    _Float16* bWcp = (_Float16*)(ws + off); off = align256(off + (size_t)2 * Hn * Hn * 2);
    _Float16* Wfp  = (_Float16*)(ws + off); off = align256(off + (size_t)Hn * Cpad * 2);

    const int M = Bn * Tn;
    pad_inputs_kernel<<<(M * Kpad1 + 255) / 256, 256, 0, stream>>>(inputs, Xpad);
#define PACK(src, dst, Ks, Ns, Kp, Np)                                                     \
    pack_weight_f16<<<(int)(((size_t)(Kp) * (Np) + 255) / 256), 256, 0, stream>>>(         \
        src, dst, Ks, Ns, Kp, Np)
    PACK(W1, W1p, Fn, Hn, Kpad1, Hn);
    PACK(W2, W2p, Hn, Hn, Hn, Hn);
    PACK(fWg, fWgp, 2 * Hn, 2 * Hn, 2 * Hn, 2 * Hn);
    PACK(fWc, fWcp, 2 * Hn, Hn, 2 * Hn, Hn);
    PACK(bWg, bWgp, 2 * Hn, 2 * Hn, 2 * Hn, 2 * Hn);
    PACK(bWc, bWcp, 2 * Hn, Hn, 2 * Hn, Hn);
    PACK(Wf, Wfp, Hn, Cn, Hn, Cpad);
#undef PACK

    {
        int tiles = (M / 16) * (Hn / 16);
        gemm_f16packed<<<(tiles + 7) / 8, 256, 0, stream>>>(Xpad, W1p, b1, h1, M, Hn, Kpad1, 1);
        gemm_f16packed<<<(tiles + 7) / 8, 256, 0, stream>>>(h1, W2p, b2, h2, M, Hn, Hn, 1);
    }
    size_t gru_lds = (size_t)Bn * Hn * 16 + Bn * 4; // h32+h16+rh+u+xstage + lens
    gru_kernel<<<2, 512, gru_lds, stream>>>(h2, fWgp, fbg, fWcp, fbc, bWgp, bbg, bWcp, bbc,
                                            seq_lens, h1, bwdbuf);
    {
        int tiles = (M / 16) * (Cpad / 16);
        proj_logits<<<(tiles + 7) / 8, 256, 0, stream>>>(h1, bwdbuf, Wfp, bfv, LP);
    }
    log_softmax_kernel<<<(Tn * Bn + 255) / 256, 256, 0, stream>>>(LP);
    ctc_kernel<<<Bn, 192, 0, stream>>>(LP, labels, seq_lens, lab_lens, ctcbuf);
    finalize_kernel<<<1, 256, 0, stream>>>(W1, b1, W2, b2, fWg, fbg, fWc, fbc, bWg, bbg,
                                           bWc, bbc, Wf, bfv, ctcbuf, (float*)d_out);
}
